// Head_34273839022820
// MI455X (gfx1250) — matrix operand order repaired and checked by reading, not compiled, not run on hardware
//
#include <hip/hip_runtime.h>

typedef __attribute__((ext_vector_type(16))) _Float16 v16h;
typedef __attribute__((ext_vector_type(8)))  float    v8f;
typedef __attribute__((ext_vector_type(4)))  unsigned v4u;
typedef __attribute__((ext_vector_type(8)))  int      v8i;
typedef __attribute__((ext_vector_type(4)))  int      v4i;

#define Bc 512
#define Tc 256
#define Cc 384
#define Hc 64

union V16H { unsigned u[8]; v16h h; };

__device__ inline v8f wmma16(v16h a, v16h b, v8f c) {
    return __builtin_amdgcn_wmma_f32_16x16x32_f16(false, a, false, b, (short)0, c, false, false);
}

__device__ inline float rmax16(float v) {
    v = fmaxf(v, __shfl_xor(v, 8, 32));
    v = fmaxf(v, __shfl_xor(v, 4, 32));
    v = fmaxf(v, __shfl_xor(v, 2, 32));
    v = fmaxf(v, __shfl_xor(v, 1, 32));
    return v;
}
__device__ inline float rsum16(float v) {
    v += __shfl_xor(v, 8, 32);
    v += __shfl_xor(v, 4, 32);
    v += __shfl_xor(v, 2, 32);
    v += __shfl_xor(v, 1, 32);
    return v;
}

// TDM: DMA a 2D f32 tile (rows x rowElems, row stride strideElems) from global
// into LDS at lds_addr. D# layout per CDNA5 ISA ch.8 (group0 + group1, groups
// 2/3 zero => 2D tensor). Issue once per wave; tracked by TENSORcnt.
__device__ inline void tdm_load_2d_f32(unsigned lds_addr, const void* gptr,
                                       unsigned rowElems, unsigned rows,
                                       unsigned strideElems) {
    unsigned long long ga = (unsigned long long)(uintptr_t)gptr;
    v4u g0;
    g0[0] = 1u;                                        // count=1, user D#
    g0[1] = lds_addr;                                  // lds_addr [63:32]
    g0[2] = (unsigned)(ga & 0xFFFFFFFFu);              // global_addr[31:0]
    g0[3] = (unsigned)((ga >> 32) & 0x1FFFFFFu)        // global_addr[56:32]
            | (2u << 30);                              // type=2 ("image")
    v8i g1;
    g1[0] = (int)(2u << 16);                           // data_size=4B
    g1[1] = (int)(rowElems << 16);                     // tensor_dim0[15:0] @ 63:48
    g1[2] = (int)((rowElems >> 16) | (rows << 16));    // dim0 hi | tensor_dim1 lo
    g1[3] = (int)(rowElems << 16);                     // dim1 hi(0) | tile_dim0 @127:112
    g1[4] = (int)rows;                                 // tile_dim1 @143:128
    g1[5] = (int)strideElems;                          // tensor_dim0_stride[31:0]
    g1[6] = 0;                                         // stride hi | dim1_stride lo
    g1[7] = 0;
    v4i z = {0, 0, 0, 0};
#if __clang_major__ >= 23
    v8i z8 = {0, 0, 0, 0, 0, 0, 0, 0};
    __builtin_amdgcn_tensor_load_to_lds(g0, g1, z, z, z8, 0);
#else
    __builtin_amdgcn_tensor_load_to_lds(g0, g1, z, z, 0);
#endif
}

// Kernel 0: W [C,H] f32 -> WT [3][H,C] f16 (k,q,v order)
__global__ void wt_kernel(const float* __restrict__ Wk, const float* __restrict__ Wq,
                          const float* __restrict__ Wv, _Float16* __restrict__ WT) {
    int idx = blockIdx.x * blockDim.x + threadIdx.x;
    if (idx >= 3 * Cc * Hc) return;
    int m   = idx / (Cc * Hc);
    int rem = idx - m * (Cc * Hc);
    int c = rem / Hc, h = rem - c * Hc;
    const float* W = (m == 0) ? Wk : ((m == 1) ? Wq : Wv);
    WT[(size_t)m * Hc * Cc + (size_t)h * Cc + c] = (_Float16)W[rem];
}

// Kernel 1: QKV projection. 3 waves/block: wave0->K, wave1->Q, wave2->V(transposed).
__global__ __launch_bounds__(96) void qkv_proj(const float* __restrict__ x,
                                               const _Float16* __restrict__ WT,
                                               _Float16* __restrict__ kf,
                                               _Float16* __restrict__ qf,
                                               _Float16* __restrict__ vT) {
    __shared__ float xs[16 * Cc];                 // 24 KB x tile
    const int blk = blockIdx.x;
    const int b  = blk >> 4;                      // T/16 == 16 tiles
    const int t0 = (blk & 15) * 16;
    const int tid = threadIdx.x;
    const int wv = __builtin_amdgcn_readfirstlane(tid) >> 5;  // scalar wave id

    // Stage x tile (contiguous 16*C floats) via the Tensor Data Mover.
    if (wv == 0) {
        tdm_load_2d_f32((unsigned)(uintptr_t)&xs[0],
                        x + ((size_t)b * Tc + t0) * Cc, Cc, 16, Cc);
        __builtin_amdgcn_s_wait_tensorcnt(0);
    }
    __syncthreads();

    const int lane = tid & 31;
    const int half = lane >> 4;
    const int ln   = lane & 15;
    const _Float16* W = WT + (size_t)wv * Hc * Cc;

    v8f acc[4] = {};                              // 16x64 output tile
    for (int k0 = 0; k0 < Cc; k0 += 32) {
        V16H a;                                   // A: x rows, f32->f16
#pragma unroll
        for (int v = 0; v < 8; v++) {
            int kfe = k0 + (v >> 2) * 16 + half * 8 + (v & 3) * 2;
            float2 f = *(const float2*)&xs[ln * Cc + kfe];
            a.h[2 * v]     = (_Float16)f.x;
            a.h[2 * v + 1] = (_Float16)f.y;
        }
#pragma unroll
        for (int n = 0; n < 4; n++) {
            V16H bm;                              // B: WT column slice, contiguous pairs
            const unsigned* wp = (const unsigned*)(W + (size_t)(n * 16 + ln) * Cc);
#pragma unroll
            for (int v = 0; v < 8; v++) bm.u[v] = wp[(k0 + (v >> 2) * 16 + half * 8 + (v & 3) * 2) >> 1];
            acc[n] = wmma16(a.h, bm.h, acc[n]);
        }
    }

    // Store epilogue: scalar-uniform branch on wave id, straight-line stores.
    const size_t bT = (size_t)b * Tc;
    if (wv == 2) {                                // V, transposed [H,T]
        _Float16* base = vT + (size_t)b * Hc * Tc;
#pragma unroll
        for (int n = 0; n < 4; n++)
#pragma unroll
            for (int r = 0; r < 8; r++)
                base[(size_t)(n * 16 + ln) * Tc + t0 + r + 8 * half] =
                    (_Float16)acc[n][r];
    } else {                                      // K or Q, [T,H]
        _Float16* base = ((wv == 0) ? kf : qf) + bT * Hc;
#pragma unroll
        for (int n = 0; n < 4; n++)
#pragma unroll
            for (int r = 0; r < 8; r++)
                base[(size_t)(t0 + r + 8 * half) * Hc + n * 16 + ln] =
                    (_Float16)acc[n][r];
    }
}

// Kernel 2: flash-style causal attention, one wave per (batch, 16-row q tile).
__global__ __launch_bounds__(32) void attn(const _Float16* __restrict__ qf,
                                           const _Float16* __restrict__ kf,
                                           const _Float16* __restrict__ vT,
                                           float* __restrict__ out) {
    __shared__ _Float16 pbuf[16 * 32];            // P tile relayout scratch
    const int blk = blockIdx.x;
    const int b  = blk >> 4;
    const int t0 = (blk & 15) * 16;
    const int lane = threadIdx.x & 31;
    const int half = lane >> 4;
    const int ln   = lane & 15;
    const size_t bT = (size_t)b * Tc;

    // Q tile in A-layout, both K-steps of 32
    V16H qa[2];
    {
        const unsigned* qp = (const unsigned*)(qf + (bT + t0 + ln) * Hc);
#pragma unroll
        for (int ks = 0; ks < 2; ks++)
#pragma unroll
            for (int v = 0; v < 8; v++)
                qa[ks].u[v] = qp[(ks * 32 + (v >> 2) * 16 + half * 8 + (v & 3) * 2) >> 1];
    }

    v8f acc[4] = {};
    float mrow[8], lrow[8], alpha[8];
#pragma unroll
    for (int r = 0; r < 8; r++) { mrow[r] = -1e30f; lrow[r] = 0.f; }
    // scale = C^-0.5, folded with log2(e) so we can use exp2
    const float kScale = 0.05103103630798288f * 1.4426950408889634f;

    for (int s0 = 0; s0 <= t0; s0 += 32) {
        // prefetch next key tile while we compute on this one
        if (s0 + 32 <= t0)
            __builtin_prefetch(kf + (bT + s0 + 32 + ln) * Hc, 0, 1);

        // ---- scores: two 16x16 tiles covering 32 keys ----
        v8f c0 = {}, c1 = {};
        const unsigned* kp0 = (const unsigned*)(kf + (bT + s0 + ln) * Hc);
        const unsigned* kp1 = (const unsigned*)(kf + (bT + s0 + 16 + ln) * Hc);
#pragma unroll
        for (int ks = 0; ks < 2; ks++) {
            V16H b0, b1;
#pragma unroll
            for (int v = 0; v < 8; v++) {
                int kk = (ks * 32 + (v >> 2) * 16 + half * 8 + (v & 3) * 2) >> 1;
                b0.u[v] = kp0[kk];
                b1.u[v] = kp1[kk];
            }
            c0 = wmma16(qa[ks].h, b0.h, c0);
            c1 = wmma16(qa[ks].h, b1.h, c1);
        }

        // ---- causal mask + online softmax (rows live in lane-halves) ----
        const bool need_mask = (s0 + 31) > t0;
#pragma unroll
        for (int r = 0; r < 8; r++) {
            int t = t0 + r + 8 * half;
            float s0v = c0[r] * kScale;
            float s1v = c1[r] * kScale;
            if (need_mask) {
                if (s0 + ln > t)      s0v = -1e30f;
                if (s0 + 16 + ln > t) s1v = -1e30f;
            }
            float rm   = rmax16(fmaxf(s0v, s1v));
            float mnew = fmaxf(mrow[r], rm);
            float al   = exp2f(mrow[r] - mnew);
            mrow[r] = mnew;
            float e0 = exp2f(s0v - mnew);
            float e1 = exp2f(s1v - mnew);
            lrow[r] = lrow[r] * al + rsum16(e0 + e1);
            alpha[r] = al;
            int m = r + 8 * half;
            pbuf[m * 32 + ln]      = (_Float16)e0;
            pbuf[m * 32 + 16 + ln] = (_Float16)e1;
        }
#pragma unroll
        for (int n = 0; n < 4; n++)
#pragma unroll
            for (int r = 0; r < 8; r++) acc[n][r] *= alpha[r];

        asm volatile("s_wait_dscnt 0" ::: "memory");

        // P back out of LDS in A-layout (16x32 f16)
        V16H pa;
#pragma unroll
        for (int v = 0; v < 8; v++) {
            int kk = (v >> 2) * 16 + half * 8 + (v & 3) * 2;
            pa.u[v] = *(const unsigned*)&pbuf[ln * 32 + kk];
        }

        // ---- acc += P @ V  (V transposed -> contiguous K pairs) ----
#pragma unroll
        for (int n = 0; n < 4; n++) {
            V16H bv;
            const unsigned* vp =
                (const unsigned*)(vT + ((size_t)b * Hc + n * 16 + ln) * Tc + s0);
#pragma unroll
            for (int v = 0; v < 8; v++) bv.u[v] = vp[((v >> 2) * 16 + half * 8 + (v & 3) * 2) >> 1];
            acc[n] = wmma16(pa.h, bv.h, acc[n]);
        }
    }

    // ---- normalize + store fp32 ----
#pragma unroll
    for (int n = 0; n < 4; n++)
#pragma unroll
        for (int r = 0; r < 8; r++) {
            int t = t0 + r + 8 * half;
            out[(bT + t) * Hc + n * 16 + ln] = acc[n][r] / lrow[r];
        }
}

extern "C" void kernel_launch(void* const* d_in, const int* in_sizes, int n_in,
                              void* d_out, int out_size, void* d_ws, size_t ws_size,
                              hipStream_t stream) {
    const float* x  = (const float*)d_in[0];
    const float* Wk = (const float*)d_in[1];
    const float* Wq = (const float*)d_in[2];
    const float* Wv = (const float*)d_in[3];
    float* out = (float*)d_out;

    const size_t qkvElems = (size_t)Bc * Tc * Hc;   // 8M halves each
    _Float16* kf = (_Float16*)d_ws;
    _Float16* qf = kf + qkvElems;
    _Float16* vT = qf + qkvElems;
    _Float16* WT = vT + qkvElems;                   // 3*C*H halves

    int nW = 3 * Cc * Hc;
    wt_kernel<<<(nW + 255) / 256, 256, 0, stream>>>(Wk, Wq, Wv, WT);
    qkv_proj<<<Bc * (Tc / 16), 96, 0, stream>>>(x, WT, kf, qf, vT);
    attn<<<Bc * (Tc / 16), 32, 0, stream>>>(qf, kf, vT, out);
}
